// PCBAM_9363028706109
// MI455X (gfx1250) — compile-verified, matrix operand order checked by reading, not run on hardware
//
#include <hip/hip_runtime.h>

typedef __attribute__((ext_vector_type(16))) _Float16 v16h;
typedef __attribute__((ext_vector_type(8)))  _Float16 v8h;
typedef __attribute__((ext_vector_type(8)))  float    v8f;
typedef __attribute__((ext_vector_type(2)))  float    v2f;

constexpr int kB = 8, kC = 64, kCr = 8, kN = 4096; // H*W = 4096

// ---------------- kernel 1: per-(b,c) mean & max over HW ----------------
__global__ __launch_bounds__(256) void k_stats(const float* __restrict__ x,
                                               float* __restrict__ avg,
                                               float* __restrict__ mx) {
  const int bc = blockIdx.x;                 // 0..B*C-1
  const float* p = x + (size_t)bc * kN;
  float s = 0.f, m = -3.0e38f;
  for (int i = threadIdx.x; i < kN; i += 256) {
    float v = p[i];
    s += v; m = fmaxf(m, v);
  }
  __shared__ float ss[256], sm[256];
  ss[threadIdx.x] = s; sm[threadIdx.x] = m;
  __syncthreads();
  for (int st = 128; st > 0; st >>= 1) {
    if (threadIdx.x < st) {
      ss[threadIdx.x] += ss[threadIdx.x + st];
      sm[threadIdx.x] = fmaxf(sm[threadIdx.x], sm[threadIdx.x + st]);
    }
    __syncthreads();
  }
  if (threadIdx.x == 0) { avg[bc] = ss[0] * (1.f / kN); mx[bc] = sm[0]; }
}

// ---------------- kernel 2: channel-attention MLP -> ca[B][C] ----------------
__global__ __launch_bounds__(64) void k_ca(const float* __restrict__ avg,
                                           const float* __restrict__ mx,
                                           const float* __restrict__ fc1,   // [Cr][C]
                                           const float* __restrict__ fc2,   // [C][Cr]
                                           float* __restrict__ ca) {
  const int b = blockIdx.x, c = threadIdx.x;
  __shared__ float av[kC], mv[kC], ha[kCr], hm[kCr];
  av[c] = avg[b * kC + c]; mv[c] = mx[b * kC + c];
  __syncthreads();
  if (c < kCr) {
    float sa = 0.f, sm = 0.f;
    for (int cc = 0; cc < kC; ++cc) {
      float w = fc1[c * kC + cc];
      sa += w * av[cc]; sm += w * mv[cc];
    }
    ha[c] = fmaxf(sa, 0.f); hm[c] = fmaxf(sm, 0.f);
  }
  __syncthreads();
  float s = 0.f;
  for (int r = 0; r < kCr; ++r) s += fc2[c * kCr + r] * (ha[r] + hm[r]);
  ca[b * kC + c] = 1.f / (1.f + __expf(-s));
}

// ---------------- kernel 3: x_c = x*ca ; channel mean/max -> sa_in ----------------
__global__ __launch_bounds__(256) void k_xc(const float* __restrict__ x,
                                            const float* __restrict__ ca,
                                            float* __restrict__ xc,
                                            float* __restrict__ sain) {
  const int b = blockIdx.y;
  const int n = blockIdx.x * 256 + threadIdx.x;
  __shared__ float cav[kC];
  if (threadIdx.x < kC) cav[threadIdx.x] = ca[b * kC + threadIdx.x];
  __syncthreads();
  const float* xb = x + (size_t)b * kC * kN;
  float* xcb = xc + (size_t)b * kC * kN;
  float s = 0.f, m = -3.0e38f;
  for (int c = 0; c < kC; ++c) {
    float v = xb[(size_t)c * kN + n] * cav[c];
    xcb[(size_t)c * kN + n] = v;
    s += v; m = fmaxf(m, v);
  }
  sain[((size_t)b * 2 + 0) * kN + n] = s * (1.f / kC);
  sain[((size_t)b * 2 + 1) * kN + n] = m;
}

// ---------------- kernel 4: 7x7 conv (pad 3) -> sigmoid gate; x_s in-place ----------------
__global__ __launch_bounds__(256) void k_sa(const float* __restrict__ sain,
                                            const float* __restrict__ saw,  // [1][2][7][7]
                                            float* __restrict__ xcs) {      // in: x_c, out: x_s
  const int b = blockIdx.y;
  const int n = blockIdx.x * 256 + threadIdx.x;
  const int h = n >> 6, w = n & 63;
  __shared__ float wts[98];
  if (threadIdx.x < 98) wts[threadIdx.x] = saw[threadIdx.x];
  __syncthreads();
  const float* s0 = sain + (size_t)b * 2 * kN;
  float acc = 0.f;
  for (int ci = 0; ci < 2; ++ci)
    for (int kh = 0; kh < 7; ++kh) {
      int hh = h + kh - 3;
      if (hh < 0 || hh >= 64) continue;
      for (int kw = 0; kw < 7; ++kw) {
        int ww = w + kw - 3;
        if (ww < 0 || ww >= 64) continue;
        acc += wts[ci * 49 + kh * 7 + kw] * s0[(size_t)ci * kN + hh * 64 + ww];
      }
    }
  const float sig = 1.f / (1.f + __expf(-acc));
  float* p = xcs + (size_t)b * kC * kN + n;
  for (int c = 0; c < kC; ++c) p[(size_t)c * kN] *= sig;
}

// ---------------- kernel 5: Q/K projections, stored transposed [B][N][8] f32 ----------------
__global__ __launch_bounds__(256) void k_qk(const float* __restrict__ x,
                                            const float* __restrict__ qw, const float* __restrict__ qb,
                                            const float* __restrict__ kw, const float* __restrict__ kb2,
                                            float* __restrict__ qT, float* __restrict__ kT) {
  const int b = blockIdx.y;
  const int n = blockIdx.x * 256 + threadIdx.x;
  __shared__ float sqw[kCr][kC], skw[kCr][kC];
  for (int i = threadIdx.x; i < kCr * kC; i += 256) {
    sqw[i >> 6][i & 63] = qw[i];
    skw[i >> 6][i & 63] = kw[i];
  }
  __syncthreads();
  const float* xb = x + (size_t)b * kC * kN + n;
  float qa[kCr], ka[kCr];
#pragma unroll
  for (int r = 0; r < kCr; ++r) { qa[r] = qb[r]; ka[r] = kb2[r]; }
  for (int c = 0; c < kC; ++c) {
    float v = xb[(size_t)c * kN];
#pragma unroll
    for (int r = 0; r < kCr; ++r) { qa[r] += sqw[r][c] * v; ka[r] += skw[r][c] * v; }
  }
  float* qo = qT + ((size_t)b * kN + n) * kCr;
  float* ko = kT + ((size_t)b * kN + n) * kCr;
#pragma unroll
  for (int r = 0; r < kCr; ++r) { qo[r] = qa[r]; ko[r] = ka[r]; }
}

// ---------------- kernel 6: V projection, stored [B][C][N] f16 ----------------
__global__ __launch_bounds__(256) void k_v(const float* __restrict__ x,
                                           const float* __restrict__ vw,   // [C][C]
                                           const float* __restrict__ vb,
                                           _Float16* __restrict__ vh) {
  const int b = blockIdx.y;
  const int n = blockIdx.x * 256 + threadIdx.x;
  __shared__ float w[kC][kC];
  for (int i = threadIdx.x; i < kC * kC; i += 256) w[i >> 6][i & 63] = vw[i];
  __syncthreads();
  const float* xb = x + (size_t)b * kC * kN + n;
  for (int g = 0; g < 4; ++g) {
    float acc[16];
#pragma unroll
    for (int j = 0; j < 16; ++j) acc[j] = vb[g * 16 + j];
    for (int c = 0; c < kC; ++c) {
      float v = xb[(size_t)c * kN];
#pragma unroll
      for (int j = 0; j < 16; ++j) acc[j] += w[g * 16 + j][c] * v;
    }
#pragma unroll
    for (int j = 0; j < 16; ++j)
      vh[((size_t)b * kC + g * 16 + j) * kN + n] = (_Float16)acc[j];
  }
}

// -------- lane helpers: butterfly max within each 16-lane row (v_permlane16_b32) --------
__device__ __forceinline__ float permxor16(float v, int s0, int s1) {
  int i = __float_as_int(v);
  int r = __builtin_amdgcn_permlane16(i, i, s0, s1, false, false);
  return __int_as_float(r);
}
__device__ __forceinline__ float rowmax16(float v) {
  v = fmaxf(v, permxor16(v, (int)0x67452301u, (int)0xEFCDAB89u)); // lane ^ 1
  v = fmaxf(v, permxor16(v, (int)0x54761032u, (int)0xDCFE98BAu)); // lane ^ 2
  v = fmaxf(v, permxor16(v, (int)0x32107654u, (int)0xBA98FEDCu)); // lane ^ 4
  v = fmaxf(v, permxor16(v, (int)0xFEDCBA98u, (int)0x76543210u)); // lane ^ 8
  return v;
}

// ---------------- kernel 7: flash-attention PAM + fused epilogue ----------------
// One wave per 16-query tile; 8 waves (128 queries) per block; grid (N/128, B).
// S = Q K^T via f32 16x16x4 WMMA (Cr=8 -> two chained K=4 steps, no padding).
// P V and the row-sum l (ones-column trick) via f16 16x16x32 WMMA.
// out = x_s + x + gamma * softmax(Q K^T) V
__global__ __launch_bounds__(256) void k_flash(
    const float* __restrict__ qT, const float* __restrict__ kT,
    const _Float16* __restrict__ vh, const float* __restrict__ xs,
    const float* __restrict__ xg, const float* __restrict__ gamma,
    float* __restrict__ out) {
  __shared__ __align__(16) _Float16 Pbuf[8][16][32];  // per-wave P relayout buffer
  const int lane = threadIdx.x & 31;
  const int wv = threadIdx.x >> 5;
  const int m = lane & 15;       // row (A) or column (B/D) index
  const int hi = lane >> 4;      // lane half selects K sub-range in fragment packing
  const int b = blockIdx.y;
  const int qbase = blockIdx.x * 128 + wv * 16;

  // Q A-fragments for f32 16x16x4: elem i = Q[m][koff + 2*hi + i]
  const float* qrow = qT + ((size_t)b * kN + qbase + m) * kCr + hi * 2;
  const v2f qa0 = *(const v2f*)(qrow);      // K (=r) 0..3
  const v2f qa1 = *(const v2f*)(qrow + 4);  // K (=r) 4..7

  v16h onesf;                                // all-ones B matrix -> row sums
#pragma unroll
  for (int i = 0; i < 16; ++i) onesf[i] = (_Float16)1.f;

  v8f o[4], lw;
#pragma unroll
  for (int g = 0; g < 8; ++g) {
    o[0][g] = 0.f; o[1][g] = 0.f; o[2][g] = 0.f; o[3][g] = 0.f; lw[g] = 0.f;
  }
  float mrow[8];
#pragma unroll
  for (int g = 0; g < 8; ++g) mrow[g] = -3.0e38f;

  const float* kbp = kT + (size_t)b * kN * kCr;
  const _Float16* vb = vh + (size_t)b * kC * kN;

  for (int j0 = 0; j0 < kN; j0 += 32) {
    // K B-fragments (f32, K=4): elem i = K'[2*hi + i][key]
    const float* kr0 = kbp + (size_t)(j0 + m) * kCr + hi * 2;
    const float* kr1 = kbp + (size_t)(j0 + 16 + m) * kCr + hi * 2;
    const v2f kb0a = *(const v2f*)(kr0);
    const v2f kb0b = *(const v2f*)(kr0 + 4);
    const v2f kb1a = *(const v2f*)(kr1);
    const v2f kb1b = *(const v2f*)(kr1 + 4);

    v8f zacc;
#pragma unroll
    for (int g = 0; g < 8; ++g) zacc[g] = 0.f;
    v8f s0 = __builtin_amdgcn_wmma_f32_16x16x4_f32(false, qa0, false, kb0a, (short)0, zacc, false, false);
    s0     = __builtin_amdgcn_wmma_f32_16x16x4_f32(false, qa1, false, kb0b, (short)0, s0,   false, false);
    v8f s1 = __builtin_amdgcn_wmma_f32_16x16x4_f32(false, qa0, false, kb1a, (short)0, zacc, false, false);
    s1     = __builtin_amdgcn_wmma_f32_16x16x4_f32(false, qa1, false, kb1b, (short)0, s1,   false, false);

    // online softmax (D-layout: row = g + 8*hi, col = m); row max via permlane16
    float sc[8];
#pragma unroll
    for (int g = 0; g < 8; ++g) {
      const float tm = rowmax16(fmaxf(s0[g], s1[g]));
      const float mn = fmaxf(mrow[g], tm);
      const float e0 = __expf(s0[g] - mn);
      const float e1 = __expf(s1[g] - mn);
      sc[g] = __expf(mrow[g] - mn);
      mrow[g] = mn;
      Pbuf[wv][g + 8 * hi][m] = (_Float16)e0;       // P row-major [16 rows][32 keys]
      Pbuf[wv][g + 8 * hi][16 + m] = (_Float16)e1;
    }
#pragma unroll
    for (int g = 0; g < 8; ++g) {
      o[0][g] *= sc[g]; o[1][g] *= sc[g]; o[2][g] *= sc[g]; o[3][g] *= sc[g];
      lw[g] *= sc[g];
    }

    // P A-fragment readback: elem i -> K = (i/8)*16 + 8*hi + i%8, row = m
    const _Float16* prow = &Pbuf[wv][m][0];
    const v8h plo = *(const v8h*)(prow + hi * 8);
    const v8h phi = *(const v8h*)(prow + 16 + hi * 8);
    const v16h pa = __builtin_shufflevector(plo, phi, 0, 1, 2, 3, 4, 5, 6, 7,
                                            8, 9, 10, 11, 12, 13, 14, 15);

    // l += P * ones  (row sums, replicated across lanes by the D layout)
    lw = __builtin_amdgcn_wmma_f32_16x16x32_f16(false, pa, false, onesf, (short)0, lw, false, false);

    // O += P[16x32] * V'[32x16] for 4 channel tiles; V' B-fragment from [C][N] f16
#pragma unroll
    for (int ct = 0; ct < 4; ++ct) {
      const _Float16* vr = vb + (size_t)(ct * 16 + m) * kN + j0;
      const v8h vlo = *(const v8h*)(vr + hi * 8);
      const v8h vhi = *(const v8h*)(vr + 16 + hi * 8);
      const v16h vf = __builtin_shufflevector(vlo, vhi, 0, 1, 2, 3, 4, 5, 6, 7,
                                              8, 9, 10, 11, 12, 13, 14, 15);
      o[ct] = __builtin_amdgcn_wmma_f32_16x16x32_f16(false, pa, false, vf, (short)0, o[ct], false, false);
    }
  }

  // epilogue: out = x_s + x + gamma * O / l  (D-layout: row i0+g, col = channel)
  const float g0 = gamma[0];
  const int i0 = qbase + 8 * hi;
#pragma unroll
  for (int ct = 0; ct < 4; ++ct) {
    const int c = ct * 16 + m;
    const size_t base = ((size_t)b * kC + c) * kN + i0;
#pragma unroll
    for (int g = 0; g < 8; ++g)
      out[base + g] = xs[base + g] + xg[base + g] + g0 * (o[ct][g] / lw[g]);
  }
}

extern "C" void kernel_launch(void* const* d_in, const int* in_sizes, int n_in,
                              void* d_out, int out_size, void* d_ws, size_t ws_size,
                              hipStream_t stream) {
  (void)in_sizes; (void)n_in; (void)out_size; (void)ws_size;
  const float* x     = (const float*)d_in[0];
  const float* fc1   = (const float*)d_in[1];
  const float* fc2   = (const float*)d_in[2];
  const float* qw    = (const float*)d_in[3];
  const float* qb    = (const float*)d_in[4];
  const float* kw    = (const float*)d_in[5];
  const float* kb    = (const float*)d_in[6];
  const float* vw    = (const float*)d_in[7];
  const float* vbias = (const float*)d_in[8];
  const float* gamma = (const float*)d_in[9];
  const float* saw   = (const float*)d_in[10];
  float* out = (float*)d_out;

  char* p = (char*)d_ws;
  auto take = [&](size_t bytes) { char* r = p; p += (bytes + 255) & ~(size_t)255; return r; };
  float*    ws_avg  = (float*)take((size_t)kB * kC * sizeof(float));
  float*    ws_mx   = (float*)take((size_t)kB * kC * sizeof(float));
  float*    ws_ca   = (float*)take((size_t)kB * kC * sizeof(float));
  float*    ws_sain = (float*)take((size_t)kB * 2 * kN * sizeof(float));
  float*    ws_xcs  = (float*)take((size_t)kB * kC * kN * sizeof(float));       // x_c then x_s (8 MB)
  float*    ws_qT   = (float*)take((size_t)kB * kN * kCr * sizeof(float));      // 1 MB
  float*    ws_kT   = (float*)take((size_t)kB * kN * kCr * sizeof(float));      // 1 MB
  _Float16* ws_v    = (_Float16*)take((size_t)kB * kC * kN * sizeof(_Float16)); // 4 MB

  k_stats<<<kB * kC, 256, 0, stream>>>(x, ws_avg, ws_mx);
  k_ca<<<kB, 64, 0, stream>>>(ws_avg, ws_mx, fc1, fc2, ws_ca);
  k_xc<<<dim3(kN / 256, kB), 256, 0, stream>>>(x, ws_ca, ws_xcs, ws_sain);
  k_sa<<<dim3(kN / 256, kB), 256, 0, stream>>>(ws_sain, saw, ws_xcs);
  k_qk<<<dim3(kN / 256, kB), 256, 0, stream>>>(x, qw, qb, kw, kb, ws_qT, ws_kT);
  k_v<<<dim3(kN / 256, kB), 256, 0, stream>>>(x, vw, vbias, ws_v);
  k_flash<<<dim3(kN / 128, kB), 256, 0, stream>>>(ws_qT, ws_kT, ws_v, ws_xcs, x, gamma, out);
}
